// CrossAttLocalBatch_83434034692458
// MI455X (gfx1250) — matrix operand order repaired and checked by reading, not compiled, not run on hardware
//
#include <hip/hip_runtime.h>

typedef __attribute__((ext_vector_type(16))) _Float16 v16h;
typedef __attribute__((ext_vector_type(8)))  float    v8f;
typedef _Float16 half_t;

union FragU { v16h h; uint4 u[2]; };

__device__ __forceinline__ float wave_sum(float v) {
#pragma unroll
  for (int m = 16; m >= 1; m >>= 1) v += __shfl_xor(v, m, 32);
  return v;
}

// ---------------- elementwise f32 -> f16 ----------------
__global__ void k_cvt_f16(const float* __restrict__ src, half_t* __restrict__ dst, int n) {
  int i = blockIdx.x * blockDim.x + threadIdx.x;
  if (i < n) dst[i] = (half_t)src[i];
}

// W [768,256] f32 -> WT [256,768] f16
__global__ void k_transpose_w(const float* __restrict__ Wm, half_t* __restrict__ WT) {
  int i = blockIdx.x * blockDim.x + threadIdx.x;   // over 768*256
  int kk = i / 256, n = i - kk * 256;
  WT[n * 768 + kk] = (half_t)Wm[i];
}

// v [48*96,256] f32 -> vT [48][256][96] f16
__global__ void k_vT(const float* __restrict__ V, half_t* __restrict__ VT) {
  int i = blockIdx.x * blockDim.x + threadIdx.x;   // over 48*96*256
  int b = i / (96 * 256);
  int r = i - b * 96 * 256;
  int j = r / 256, d = r - j * 256;
  VT[((size_t)b * 256 + d) * 96 + j] = (half_t)V[i];
}

// ---------------- QKV projection: [4608,768] @ [768,256] ----------------
// One wave per 16x16 output tile, K=768 in 24 WMMA steps.
__global__ __launch_bounds__(256)
void k_gemm_qkv(const half_t* __restrict__ a16, const half_t* __restrict__ b16,
                const half_t* __restrict__ wqT, const half_t* __restrict__ wkT,
                const half_t* __restrict__ wvT,
                const float* __restrict__ bq, const float* __restrict__ bk,
                const float* __restrict__ bv,
                float* __restrict__ Q, float* __restrict__ K, float* __restrict__ V) {
  const int z = blockIdx.y;
  const half_t* A    = (z == 0) ? a16 : b16;
  const half_t* WT   = (z == 0) ? wqT : (z == 1) ? wkT : wvT;
  const float*  bias = (z == 0) ? bq  : (z == 1) ? bk  : bv;
  float*        OUT  = (z == 0) ? Q   : (z == 1) ? K   : V;

  const int tid = threadIdx.x;
  const int w = tid >> 5, lane = tid & 31;
  const int kg = lane >> 4, ml = lane & 15;

  const int tile = blockIdx.x * 8 + w;   // 0..4607
  const int tm = tile >> 4;              // M tile (0..287)
  const int tn = tile & 15;              // N tile (0..15)

  const half_t* arow = A  + (size_t)(tm * 16 + ml) * 768;
  const half_t* brow = WT + (size_t)(tn * 16 + ml) * 768;

  v8f acc = {};
#pragma unroll
  for (int ks = 0; ks < 24; ++ks) {
    const int k0 = ks * 32;
    FragU fa, fb;
    fa.u[0] = *(const uint4*)(arow + k0 + kg * 8);
    fa.u[1] = *(const uint4*)(arow + k0 + 16 + kg * 8);
    fb.u[0] = *(const uint4*)(brow + k0 + kg * 8);
    fb.u[1] = *(const uint4*)(brow + k0 + 16 + kg * 8);
    acc = __builtin_amdgcn_wmma_f32_16x16x32_f16(false, fa.h, false, fb.h,
                                                 (short)0, acc, false, false);
  }
  const int n  = tn * 16 + ml;
  const int m0 = tm * 16 + (lane < 16 ? 0 : 8);
  const float bval = bias[n];
#pragma unroll
  for (int r = 0; r < 8; ++r)
    OUT[(size_t)(m0 + r) * 256 + n] = acc[r] + bval;
}

// ---------------- per-row L2 normalize (cosine), f32 -> f16 ----------------
__global__ void k_rownorm(const float* __restrict__ X, half_t* __restrict__ XN) {
  const int w = threadIdx.x >> 5, lane = threadIdx.x & 31;
  const int row = blockIdx.x * 8 + w;     // 0..4607
  const float* x = X + (size_t)row * 256;
  float v[8], ss = 0.f;
#pragma unroll
  for (int t = 0; t < 8; ++t) { v[t] = x[lane + 32 * t]; ss += v[t] * v[t]; }
  ss = wave_sum(ss);
  const float sc = 1.0f / fmaxf(sqrtf(ss), 1e-8f);
  half_t* o = XN + (size_t)row * 256;
#pragma unroll
  for (int t = 0; t < 8; ++t) o[lane + 32 * t] = (half_t)(v[t] * sc);
}

// ---------------- fused cosine-GEMM + Sinkhorn + T@v + residual + LayerNorm -----
// One workgroup (8 waves) per (a,b) pair; 96x96 transport plan lives in LDS.
__global__ __launch_bounds__(256)
void k_fused(const half_t* __restrict__ qn16, const half_t* __restrict__ kn16,
             const half_t* __restrict__ vT16, const float* __restrict__ Q,
             const float* __restrict__ ln_g, const float* __restrict__ ln_b,
             float* __restrict__ out) {
  __shared__ __align__(16) char smem[18432 + 36864];
  half_t* sT16 = (half_t*)smem;             // 96*96 f16 (18 KB)
  float*  sC   = (float*)(smem + 18432);    // 96*96 f32 (36 KB)
  float*  sAtt = (float*)(smem + 18432);    // 16*256 f32 (aliases sC; sC dead by then)

  const int a = blockIdx.x, b = blockIdx.y;
  const int tid = threadIdx.x;
  const int w = tid >> 5, lane = tid & 31;
  const int kg = lane >> 4, ml = lane & 15;

  const half_t* qa = qn16 + (size_t)a * 96 * 256;
  const half_t* kb = kn16 + (size_t)b * 96 * 256;
  const half_t* vb = vT16 + (size_t)b * 256 * 96;

  // ---- Phase 1: C = qn_a @ kn_b^T (96x96), f16 WMMA, f32 accum -> LDS ----
  for (int t = w; t < 36; t += 8) {
    const int tm = t / 6, tn = t - tm * 6;
    const half_t* arow = qa + (size_t)(tm * 16 + ml) * 256;
    const half_t* brow = kb + (size_t)(tn * 16 + ml) * 256;
    v8f acc = {};
#pragma unroll
    for (int ks = 0; ks < 8; ++ks) {
      const int k0 = ks * 32;
      FragU fa, fb;
      fa.u[0] = *(const uint4*)(arow + k0 + kg * 8);
      fa.u[1] = *(const uint4*)(arow + k0 + 16 + kg * 8);
      fb.u[0] = *(const uint4*)(brow + k0 + kg * 8);
      fb.u[1] = *(const uint4*)(brow + k0 + 16 + kg * 8);
      acc = __builtin_amdgcn_wmma_f32_16x16x32_f16(false, fa.h, false, fb.h,
                                                   (short)0, acc, false, false);
    }
    const int n  = tn * 16 + ml;
    const int m0 = tm * 16 + (lane < 16 ? 0 : 8);
#pragma unroll
    for (int r = 0; r < 8; ++r) sC[(m0 + r) * 96 + n] = acc[r];
  }
  __syncthreads();

  // ---- Phase 2: Sinkhorn on LDS tile: T = exp(C/0.1), 5x row/col renorm ----
  for (int i = tid; i < 9216; i += 256) sC[i] = __expf(sC[i] * 10.0f);
  __syncthreads();

  for (int it = 0; it < 5; ++it) {
    for (int t = 0; t < 12; ++t) {          // rows: 12 per wave
      const int i = w + 8 * t;
      float s = sC[i*96 + lane] + sC[i*96 + lane + 32] + sC[i*96 + lane + 64];
      s = wave_sum(s);
      const float inv = 1.0f / s;
      sC[i*96 + lane]      *= inv;
      sC[i*96 + lane + 32] *= inv;
      sC[i*96 + lane + 64] *= inv;
    }
    __syncthreads();
    for (int t = 0; t < 12; ++t) {          // cols: 12 per wave
      const int j = w + 8 * t;
      float s = sC[lane*96 + j] + sC[(lane+32)*96 + j] + sC[(lane+64)*96 + j];
      s = wave_sum(s);
      const float inv = 1.0f / s;
      sC[lane*96 + j]      *= inv;
      sC[(lane+32)*96 + j] *= inv;
      sC[(lane+64)*96 + j] *= inv;
    }
    __syncthreads();
  }

  // ---- Row L2 normalize, convert T -> f16 fragment source in LDS ----
  for (int t = 0; t < 12; ++t) {
    const int i = w + 8 * t;
    float x0 = sC[i*96 + lane], x1 = sC[i*96 + lane + 32], x2 = sC[i*96 + lane + 64];
    float ss = wave_sum(x0*x0 + x1*x1 + x2*x2);
    const float sc = 1.0f / fmaxf(sqrtf(ss), 1e-12f);
    sT16[i*96 + lane]      = (half_t)(x0 * sc);
    sT16[i*96 + lane + 32] = (half_t)(x1 * sc);
    sT16[i*96 + lane + 64] = (half_t)(x2 * sc);
  }
  __syncthreads();   // sC dead; sAtt may now reuse its space

  // ---- Phase 3: attended = T @ v + q, then LayerNorm; 16 output rows/pass ----
  const float* qbase = Q + (size_t)a * 96 * 256;
  float* outb = out + ((size_t)a * 48 + b) * 96 * 256;

  for (int tmB = 0; tmB < 6; ++tmB) {
#pragma unroll
    for (int hh = 0; hh < 2; ++hh) {
      const int tn = w + 8 * hh;            // N tile 0..15
      const half_t* arow = sT16 + (tmB * 16 + ml) * 96;       // A from LDS (ds_load_b128)
      const half_t* brow = vb + (size_t)(tn * 16 + ml) * 96;  // B from pre-transposed v
      v8f acc = {};
#pragma unroll
      for (int ks = 0; ks < 3; ++ks) {      // K = 96
        const int k0 = ks * 32;
        FragU fa, fb;
        fa.u[0] = *(const uint4*)(arow + k0 + kg * 8);
        fa.u[1] = *(const uint4*)(arow + k0 + 16 + kg * 8);
        fb.u[0] = *(const uint4*)(brow + k0 + kg * 8);
        fb.u[1] = *(const uint4*)(brow + k0 + 16 + kg * 8);
        acc = __builtin_amdgcn_wmma_f32_16x16x32_f16(false, fa.h, false, fb.h,
                                                     (short)0, acc, false, false);
      }
      const int n  = tn * 16 + ml;
      const int m0 = (lane < 16 ? 0 : 8);
#pragma unroll
      for (int r = 0; r < 8; ++r) sAtt[(m0 + r) * 256 + n] = acc[r];
    }
    __syncthreads();

    // LayerNorm over H=256 for rows tmB*16 .. tmB*16+15 (2 rows per wave)
#pragma unroll
    for (int rr = 0; rr < 2; ++rr) {
      const int iLoc = w * 2 + rr;
      const int i = tmB * 16 + iLoc;
      const float* qrow = qbase + (size_t)i * 256;
      float xv[8]; float s = 0.f, ss2 = 0.f;
#pragma unroll
      for (int t = 0; t < 8; ++t) {
        const int d = lane + 32 * t;
        float x = sAtt[iLoc * 256 + d] + qrow[d];
        xv[t] = x; s += x; ss2 += x * x;
      }
      s = wave_sum(s); ss2 = wave_sum(ss2);
      const float mu  = s * (1.0f / 256.0f);
      const float var = ss2 * (1.0f / 256.0f) - mu * mu;
      const float rs  = rsqrtf(var + 1e-5f);
      float* orow = outb + (size_t)i * 256;
#pragma unroll
      for (int t = 0; t < 8; ++t) {
        const int d = lane + 32 * t;
        orow[d] = (xv[t] - mu) * rs * ln_g[d] + ln_b[d];
      }
    }
    __syncthreads();  // sAtt reused next tmB pass
  }
}

extern "C" void kernel_launch(void* const* d_in, const int* in_sizes, int n_in,
                              void* d_out, int out_size, void* d_ws, size_t ws_size,
                              hipStream_t stream) {
  (void)in_sizes; (void)n_in; (void)out_size; (void)ws_size;
  const float* batch_a = (const float*)d_in[0];
  const float* batch_b = (const float*)d_in[1];
  const float* Wq = (const float*)d_in[2];
  const float* bq = (const float*)d_in[3];
  const float* Wk = (const float*)d_in[4];
  const float* bk = (const float*)d_in[5];
  const float* Wv = (const float*)d_in[6];
  const float* bv = (const float*)d_in[7];
  const float* ln_g = (const float*)d_in[8];
  const float* ln_b = (const float*)d_in[9];
  float* out = (float*)d_out;

  // workspace carve (~36.6 MB total)
  char* wsp = (char*)d_ws;
  auto carve = [&](size_t bytes) {
    char* p = wsp; wsp += (bytes + 255) & ~(size_t)255; return p;
  };
  half_t* a16  = (half_t*)carve(4608ull * 768 * 2);
  half_t* b16  = (half_t*)carve(4608ull * 768 * 2);
  half_t* wqT  = (half_t*)carve(256ull * 768 * 2);
  half_t* wkT  = (half_t*)carve(256ull * 768 * 2);
  half_t* wvT  = (half_t*)carve(256ull * 768 * 2);
  float*  Qf   = (float*)carve(4608ull * 256 * 4);
  float*  Kf   = (float*)carve(4608ull * 256 * 4);
  float*  Vf   = (float*)carve(4608ull * 256 * 4);
  half_t* qn16 = (half_t*)carve(4608ull * 256 * 2);
  half_t* kn16 = (half_t*)carve(4608ull * 256 * 2);
  half_t* vT16 = (half_t*)carve(48ull * 256 * 96 * 2);

  k_cvt_f16<<<13824, 256, 0, stream>>>(batch_a, a16, 4608 * 768);
  k_cvt_f16<<<13824, 256, 0, stream>>>(batch_b, b16, 4608 * 768);
  k_transpose_w<<<768, 256, 0, stream>>>(Wq, wqT);
  k_transpose_w<<<768, 256, 0, stream>>>(Wk, wkT);
  k_transpose_w<<<768, 256, 0, stream>>>(Wv, wvT);
  k_gemm_qkv<<<dim3(576, 3), 256, 0, stream>>>(a16, b16, wqT, wkT, wvT,
                                               bq, bk, bv, Qf, Kf, Vf);
  k_rownorm<<<576, 256, 0, stream>>>(Qf, qn16);
  k_rownorm<<<576, 256, 0, stream>>>(Kf, kn16);
  k_vT<<<4608, 256, 0, stream>>>(Vf, vT16);
  k_fused<<<dim3(48, 48), 256, 0, stream>>>(qn16, kn16, vT16, Qf, ln_g, ln_b, out);
}
